// TLN_18021682774663
// MI455X (gfx1250) — matrix operand order repaired and checked by reading, not compiled, not run on hardware
//
#include <hip/hip_runtime.h>
#include <math.h>

// ---------------- shapes ----------------
constexpr int Bn  = 128;
constexpr int L0  = 719;
constexpr int C1  = 16,  KW1 = 32, L1 = 688, L1P = 343;
constexpr int L2  = 171;              // after miu pool
constexpr int CM  = 32;               // x_miu channels
constexpr int CA1 = 64,  LA1 = 85;    // h1
constexpr int CA2 = 32,  LA2 = 42;    // h2
constexpr int LU1 = 84,  L3  = 86,  C3 = 64;   // h3
constexpr int LU2 = 172, L4  = 174, C4 = 32;   // h4
constexpr int CI  = 160, LI  = 171;   // inp
constexpr int LT  = 170;              // until / logic length
constexpr int CL1 = 20;               // and1/or1 out channels
constexpr int CX1 = 200;              // concat channels
constexpr int CL2 = 6;
constexpr int CX2 = 12;
constexpr int Mrows = Bn * CX2;       // 1536
constexpr int K1 = 170, Kp1 = 192, N1 = 1024;
constexpr int K2 = 1024, N2 = 256;
constexpr int K3 = 256,  N3 = 10, Np3 = 16;
constexpr float EPS = 1e-6f;

// ---------------- workspace layout (floats; every block 64B-aligned) ----------------
constexpr size_t o_conv1 = 0;
constexpr size_t o_xf    = o_conv1 + (size_t)Bn*C1*L1;
constexpr size_t o_xmiu  = o_xf    + (size_t)Bn*C1*L1P;
constexpr size_t o_h1    = o_xmiu  + (size_t)Bn*CM*L2;
constexpr size_t o_h2    = o_h1    + (size_t)Bn*CA1*LA1;
constexpr size_t o_h3    = o_h2    + (size_t)Bn*CA2*LA2;
constexpr size_t o_h4    = o_h3    + (size_t)Bn*C3*L3;
constexpr size_t o_inp   = o_h4    + (size_t)Bn*C4*L4;
constexpr size_t o_x1    = o_inp   + (size_t)Bn*CI*LI;
constexpr size_t o_x2    = o_x1    + (size_t)Bn*CX1*LT;
constexpr size_t o_wa2   = o_x2    + (size_t)Bn*CX2*LT;
constexpr size_t o_wo2   = o_wa2   + 20*320;
constexpr size_t o_wa    = o_wo2   + 20*320;
constexpr size_t o_wo    = o_wa    + 6*400;
constexpr size_t o_stat  = o_wo    + 6*400;
// f16 regions (sizes in halves / 2 -> floats); all counts are multiples of 16 floats
constexpr size_t o_apack = o_stat  + 32;                          // 1536*192 h
constexpr size_t o_w1p   = o_apack + (size_t)Mrows*Kp1/2;         // 1024*192 h
constexpr size_t o_w2p   = o_w1p   + (size_t)N1*Kp1/2;            // 256*1024 h
constexpr size_t o_w3p   = o_w2p   + (size_t)N2*K2/2;             // 16*256 h
constexpr size_t o_f1h   = o_w3p   + (size_t)Np3*K3/2;            // 1536*1024 h
constexpr size_t o_f2h   = o_f1h   + (size_t)Mrows*N1/2;          // 1536*256 h

static inline int cdiv(int a, int b) { return (a + b - 1) / b; }

__device__ __forceinline__ float sigf(float v) { return 1.0f / (1.0f + expf(-v)); }

// ---------------- conv1 (valid, 1->16, k=32) ----------------
__global__ void tln_conv1(const float* __restrict__ x, const float* __restrict__ w,
                          float* __restrict__ out, int total) {
  int i = blockIdx.x * blockDim.x + threadIdx.x;
  if (i >= total) return;
  int l = i % L1; int c = (i / L1) % C1; int b = i / (L1 * C1);
  const float* xr = x + (size_t)b * L0;
  const float* wr = w + c * KW1;
  float s = 0.f;
  #pragma unroll 8
  for (int t = 0; t < KW1; ++t) s += xr[l + t] * wr[t];
  out[i] = s;
}

// ---------------- batchnorm statistics (one block / channel) ----------------
__global__ void tln_bnstats(const float* __restrict__ y, float* __restrict__ stat) {
  __shared__ float ss[256], sq[256];
  int c = blockIdx.x;
  float s = 0.f, q = 0.f;
  for (int i = threadIdx.x; i < Bn * L1; i += blockDim.x) {
    int b = i / L1, l = i % L1;
    float v = y[((size_t)b * C1 + c) * L1 + l];
    s += v; q += v * v;
  }
  ss[threadIdx.x] = s; sq[threadIdx.x] = q;
  __syncthreads();
  for (int st = 128; st > 0; st >>= 1) {
    if (threadIdx.x < st) { ss[threadIdx.x] += ss[threadIdx.x + st]; sq[threadIdx.x] += sq[threadIdx.x + st]; }
    __syncthreads();
  }
  if (threadIdx.x == 0) {
    float n = (float)(Bn * L1);
    float m = ss[0] / n;
    stat[c] = m;
    stat[16 + c] = sq[0] / n - m * m;
  }
}

// ---------------- BN + relu + maxpool3s2 ----------------
__global__ void tln_bnpool(const float* __restrict__ y, const float* __restrict__ stat,
                           const float* __restrict__ g, const float* __restrict__ be,
                           float* __restrict__ xf, int total) {
  int i = blockIdx.x * blockDim.x + threadIdx.x;
  if (i >= total) return;
  int p = i % L1P; int c = (i / L1P) % C1; int b = i / (L1P * C1);
  float inv = rsqrtf(stat[16 + c] + 1e-5f);
  float m = stat[c], gg = g[c], bb = be[c];
  float best = -1e30f;
  for (int j = 0; j < 3; ++j) {
    float v = y[((size_t)b * C1 + c) * L1 + 2 * p + j];
    v = fmaxf(gg * (v - m) * inv + bb, 0.f);
    best = fmaxf(best, v);
  }
  xf[i] = best;
}

// ---------------- miu (pos/neg) + maxpool3s2 -> x_miu concat ----------------
__global__ void tln_miu(const float* __restrict__ xf,
                        const float* __restrict__ ap, const float* __restrict__ bp,
                        const float* __restrict__ an, const float* __restrict__ bnv,
                        float* __restrict__ xmiu, int total) {
  int i = blockIdx.x * blockDim.x + threadIdx.x;
  if (i >= total) return;
  int p = i % L2; int c = (i / L2) % C1; int b = i / (L2 * C1);
  float a1 = ap[c], b1 = bp[c], a2 = an[c], b2 = bnv[c];
  float mp = -1e30f, mn = -1e30f;
  for (int j = 0; j < 3; ++j) {
    float v = xf[((size_t)b * C1 + c) * L1P + 2 * p + j];
    mp = fmaxf(mp, sigf(a1 * (v - b1)));
    mn = fmaxf(mn, sigf(-a2 * (v - b2)));
  }
  xmiu[((size_t)b * CM + c) * L2 + p]      = mp;
  xmiu[((size_t)b * CM + 16 + c) * L2 + p] = mn;
}

// ---------------- ae conv1 (32->64,k3,pad1) + tanh + pool3s2 ----------------
__global__ void tln_ae1(const float* __restrict__ xm, const float* __restrict__ w,
                        const float* __restrict__ bias, float* __restrict__ h1, int total) {
  int i = blockIdx.x * blockDim.x + threadIdx.x;
  if (i >= total) return;
  int p = i % LA1; int o = (i / LA1) % CA1; int b = i / (LA1 * CA1);
  float best = -1e30f;
  for (int j = 0; j < 3; ++j) {
    int l = 2 * p + j;
    float s = bias[o];
    for (int c = 0; c < CM; ++c) {
      const float* xr = xm + ((size_t)b * CM + c) * L2;
      const float* wr = w + (o * CM + c) * 3;
      for (int t = 0; t < 3; ++t) {
        int pos = l - 1 + t;
        if (pos >= 0 && pos < L2) s += xr[pos] * wr[t];
      }
    }
    best = fmaxf(best, tanhf(s));
  }
  h1[i] = best;
}

// ---------------- ae conv2 (64->32,k3,pad1) + relu + pool3s2 ----------------
__global__ void tln_ae2(const float* __restrict__ h1, const float* __restrict__ w,
                        const float* __restrict__ bias, float* __restrict__ h2, int total) {
  int i = blockIdx.x * blockDim.x + threadIdx.x;
  if (i >= total) return;
  int p = i % LA2; int o = (i / LA2) % CA2; int b = i / (LA2 * CA2);
  float best = -1e30f;
  for (int j = 0; j < 3; ++j) {
    int l = 2 * p + j;
    float s = bias[o];
    for (int c = 0; c < CA1; ++c) {
      const float* xr = h1 + ((size_t)b * CA1 + c) * LA1;
      const float* wr = w + (o * CA1 + c) * 3;
      for (int t = 0; t < 3; ++t) {
        int pos = l - 1 + t;
        if (pos >= 0 && pos < LA1) s += xr[pos] * wr[t];
      }
    }
    best = fmaxf(best, fmaxf(s, 0.f));
  }
  h2[i] = best;
}

// linear resize (half-pixel centers, edge clamp), factor 2
__device__ __forceinline__ float up2(const float* row, int Lin, int j) {
  float src = 0.5f * (j + 0.5f) - 0.5f;
  float fi = floorf(src);
  int i0 = (int)fi;
  float wv = src - fi;
  int ia = min(max(i0, 0), Lin - 1);
  int ib = min(max(i0 + 1, 0), Lin - 1);
  return row[ia] * (1.f - wv) + row[ib] * wv;
}

// ---------------- upsample + ae conv3 (32->64,k3,pad2) + tanh ----------------
__global__ void tln_ae3(const float* __restrict__ h2, const float* __restrict__ w,
                        const float* __restrict__ bias, float* __restrict__ h3, int total) {
  int i = blockIdx.x * blockDim.x + threadIdx.x;
  if (i >= total) return;
  int l = i % L3; int o = (i / L3) % C3; int b = i / (L3 * C3);
  float s = bias[o];
  for (int c = 0; c < CA2; ++c) {
    const float* xr = h2 + ((size_t)b * CA2 + c) * LA2;
    const float* wr = w + (o * CA2 + c) * 3;
    for (int t = 0; t < 3; ++t) {
      int pos = l - 2 + t;
      if (pos >= 0 && pos < LU1) s += up2(xr, LA2, pos) * wr[t];
    }
  }
  h3[i] = tanhf(s);
}

// ---------------- upsample + ae conv4 (64->32,k3,pad2) + sigmoid ----------------
__global__ void tln_ae4(const float* __restrict__ h3, const float* __restrict__ w,
                        const float* __restrict__ bias, float* __restrict__ h4, int total) {
  int i = blockIdx.x * blockDim.x + threadIdx.x;
  if (i >= total) return;
  int l = i % L4; int o = (i / L4) % C4; int b = i / (L4 * C4);
  float s = bias[o];
  for (int c = 0; c < C3; ++c) {
    const float* xr = h3 + ((size_t)b * C3 + c) * L3;
    const float* wr = w + (o * C3 + c) * 3;
    for (int t = 0; t < 3; ++t) {
      int pos = l - 2 + t;
      if (pos >= 0 && pos < LU2) s += up2(xr, L3, pos) * wr[t];
    }
  }
  h4[i] = sigf(s);
}

// ---------------- maxpool4s1 + K_SEG masking -> inp (160ch) ----------------
__global__ void tln_seg(const float* __restrict__ h4, const float* __restrict__ xm,
                        float* __restrict__ inp, int total) {
  int i = blockIdx.x * blockDim.x + threadIdx.x;
  if (i >= total) return;
  int l = i % LI; int c = (i / LI) % CM; int b = i / (LI * CM);
  const float* hr = h4 + ((size_t)b * C4 + c) * L4;
  float wv = fmaxf(fmaxf(hr[l], hr[l + 1]), fmaxf(hr[l + 2], hr[l + 3]));
  float xv = xm[((size_t)b * CM + c) * L2 + l];
  for (int k = 0; k < 5; ++k) {
    float lo = 0.2f * k;
    float hi = (k == 4) ? 2.0f : 0.2f * (k + 1);
    float val = (wv >= lo && wv < hi) ? wv * xv : 0.0f;
    inp[((size_t)b * CI + k * 32 + c) * LI + l] = val;
  }
}

// ---------------- until(T=25), writes into x1 channels [40,200) ----------------
__global__ void tln_until(const float* __restrict__ inp, float* __restrict__ x1, int total) {
  int i = blockIdx.x * blockDim.x + threadIdx.x;
  if (i >= total) return;
  int l = i % LT; int c = (i / LT) % CI; int b = i / (LT * CI);
  const float* xr = inp + ((size_t)b * CI + c) * LI;
  float runmin = xr[l];
  float out = 0.f;
  #pragma unroll
  for (int d = 1; d <= 25; ++d) {
    float xs = (l + d < LI) ? xr[l + d] : 0.f;
    out = fmaxf(out, fminf(runmin, xs));
    runmin = fminf(runmin, xs);
  }
  x1[((size_t)b * CX1 + 40 + c) * LT + l] = out;
}

// ---------------- softmax of logic weights (tiny) ----------------
__global__ void tln_softmax(const float* __restrict__ a2, const float* __restrict__ o2,
                            const float* __restrict__ aw, const float* __restrict__ ow,
                            float* __restrict__ wa2, float* __restrict__ wo2,
                            float* __restrict__ wa, float* __restrict__ wo) {
  int t = threadIdx.x;
  const float* src; float* dst; int len;
  if      (t < 20) { src = a2 + t * 320;        dst = wa2 + t * 320;        len = 320; }
  else if (t < 40) { src = o2 + (t - 20) * 320; dst = wo2 + (t - 20) * 320; len = 320; }
  else if (t < 46) { src = aw + (t - 40) * 400; dst = wa  + (t - 40) * 400; len = 400; }
  else if (t < 52) { src = ow + (t - 46) * 400; dst = wo  + (t - 46) * 400; len = 400; }
  else return;
  float mx = -1e30f;
  for (int i = 0; i < len; ++i) mx = fmaxf(mx, src[i]);
  float s = 0.f;
  for (int i = 0; i < len; ++i) s += expf(src[i] - mx);
  float inv = 1.0f / s;
  for (int i = 0; i < len; ++i) dst[i] = expf(src[i] - mx) * inv;
}

// ---------------- logic layer 1 (valid, k=2) -> x1 ch [0,20) and [20,40) ----------------
__global__ void tln_logic1(const float* __restrict__ inp, const float* __restrict__ wa2,
                           const float* __restrict__ wo2, float* __restrict__ x1, int total) {
  int i = blockIdx.x * blockDim.x + threadIdx.x;
  if (i >= total) return;
  int l = i % LT; int o = (i / LT) % CL1; int b = i / (LT * CL1);
  const float* wra = wa2 + o * 320;
  const float* wro = wo2 + o * 320;
  float sA = 0.f, sO = 0.f;
  for (int c = 0; c < CI; ++c) {
    const float* xr = inp + ((size_t)b * CI + c) * LI + l;
    for (int t = 0; t < 2; ++t) {
      float xv = fminf(fmaxf(xr[t], 0.f), 1.f);
      sA += wra[c * 2 + t] * logf(xv + EPS);
      sO += wro[c * 2 + t] * logf(1.0f - xv + EPS);
    }
  }
  x1[((size_t)b * CX1 + o) * LT + l]      = fmaxf(expf(sA), 0.f);
  x1[((size_t)b * CX1 + 20 + o) * LT + l] = fmaxf(1.0f - expf(sO), 0.f);
}

// ---------------- logic layer 2 (pad (1,0), k=2) -> x2 ----------------
__global__ void tln_logic2(const float* __restrict__ x1, const float* __restrict__ wa,
                           const float* __restrict__ wo, float* __restrict__ x2, int total) {
  int i = blockIdx.x * blockDim.x + threadIdx.x;
  if (i >= total) return;
  int l = i % LT; int o = (i / LT) % CL2; int b = i / (LT * CL2);
  const float* wra = wa + o * 400;
  const float* wro = wo + o * 400;
  float sA = 0.f, sO = 0.f;
  for (int c = 0; c < CX1; ++c) {
    const float* xr = x1 + ((size_t)b * CX1 + c) * LT;
    for (int t = 0; t < 2; ++t) {
      int pos = l - 1 + t;
      float xv = (pos >= 0) ? xr[pos] : 0.f;
      xv = fminf(fmaxf(xv, 0.f), 1.f);
      sA += wra[c * 2 + t] * logf(xv + EPS);
      sO += wro[c * 2 + t] * logf(1.0f - xv + EPS);
    }
  }
  x2[((size_t)b * CX2 + o) * LT + l]     = fmaxf(expf(sA), 0.f);
  x2[((size_t)b * CX2 + 6 + o) * LT + l] = fmaxf(1.0f - expf(sO), 0.f);
}

// ---------------- f32 -> f16 pack with zero padding ----------------
__global__ void tln_pack(const float* __restrict__ src, _Float16* __restrict__ dst,
                         int rs, int cs, int cd, int total) {
  int i = blockIdx.x * blockDim.x + threadIdx.x;
  if (i >= total) return;
  int c = i % cd, r = i / cd;
  float v = (r < rs && c < cs) ? src[(size_t)r * cs + c] : 0.f;
  dst[i] = (_Float16)v;
}

// ---------------- WMMA GEMM core ----------------
typedef __attribute__((ext_vector_type(16))) _Float16 v16h;
typedef __attribute__((ext_vector_type(8)))  _Float16 v8h;
typedef __attribute__((ext_vector_type(8)))  float    v8f;

__device__ __forceinline__ v16h cat8(v8h x, v8h y) {
  return __builtin_shufflevector(x, y, 0,1,2,3,4,5,6,7,8,9,10,11,12,13,14,15);
}

// Kp must be a multiple of 64. A: (M x Kp) f16 row-major; W: (N x Kp) f16 row-major.
// One wave -> one 16x16 tile, two independent accumulator chains.
__device__ __forceinline__ v8f tln_tile_acc(const _Float16* __restrict__ A,
                                            const _Float16* __restrict__ W,
                                            int Kp, int mrow, int nrow, int laneHi) {
  const _Float16* Ar = A + (size_t)mrow * Kp + (laneHi << 3);   // +8 halves for hi lanes
  const _Float16* Wr = W + (size_t)nrow * Kp + (laneHi << 3);   // +8 halves for hi lanes
  v8f acc0 = {}; v8f acc1 = {};
  for (int kb = 0; kb < Kp; kb += 64) {
    // k-step 0: A halves [kb .. kb+7] and [kb+16 .. kb+23]; B halves [kb .. kb+7] and [kb+16 .. kb+23]
    v16h a = cat8(*(const v8h*)(Ar + kb),      *(const v8h*)(Ar + kb + 16));
    v16h b = cat8(*(const v8h*)(Wr + kb),      *(const v8h*)(Wr + kb + 16));
    acc0 = __builtin_amdgcn_wmma_f32_16x16x32_f16(false, a, false, b,
                                                  (short)0, acc0, false, false);
    // k-step 1 (kb+32), independent chain
    v16h a2 = cat8(*(const v8h*)(Ar + kb + 32), *(const v8h*)(Ar + kb + 48));
    v16h b2 = cat8(*(const v8h*)(Wr + kb + 32), *(const v8h*)(Wr + kb + 48));
    acc1 = __builtin_amdgcn_wmma_f32_16x16x32_f16(false, a2, false, b2,
                                                  (short)0, acc1, false, false);
  }
  return acc0 + acc1;
}

// D = relu(A*W^T + bias), f16 output (M x N), N multiple of 16
__global__ __launch_bounds__(256)
void tln_gemm_f16out(const _Float16* __restrict__ A, const _Float16* __restrict__ W,
                     const float* __restrict__ bias, _Float16* __restrict__ C,
                     int Kp, int N, int tilesN) {
  const int lane = threadIdx.x & 31;
  const int laneHi = lane >> 4, llo = lane & 15;
  const int tile = blockIdx.x * 8 + (threadIdx.x >> 5);
  const int mt = (tile / tilesN) << 4;
  const int nt = (tile % tilesN) << 4;
  v8f acc = tln_tile_acc(A, W, Kp, mt + llo, nt + llo, laneHi);
  float bz = bias[nt + llo];
  #pragma unroll
  for (int r = 0; r < 8; ++r) {
    float vv = fmaxf(acc[r] + bz, 0.f);
    C[(size_t)(mt + r + (laneHi << 3)) * N + nt + llo] = (_Float16)vv;
  }
}

// D = A*W^T + bias, f32 output (M x Nvalid), no relu, tail-guarded columns
__global__ __launch_bounds__(256)
void tln_gemm_f32out(const _Float16* __restrict__ A, const _Float16* __restrict__ W,
                     const float* __restrict__ bias, float* __restrict__ C,
                     int Kp, int Nvalid, int tilesN) {
  const int lane = threadIdx.x & 31;
  const int laneHi = lane >> 4, llo = lane & 15;
  const int tile = blockIdx.x * 8 + (threadIdx.x >> 5);
  const int mt = (tile / tilesN) << 4;
  const int nt = (tile % tilesN) << 4;
  v8f acc = tln_tile_acc(A, W, Kp, mt + llo, nt + llo, laneHi);
  int n = nt + llo;
  if (n < Nvalid) {
    float bz = bias[n];
    #pragma unroll
    for (int r = 0; r < 8; ++r)
      C[(size_t)(mt + r + (laneHi << 3)) * Nvalid + n] = acc[r] + bz;
  }
}

extern "C" void kernel_launch(void* const* d_in, const int* in_sizes, int n_in,
                              void* d_out, int out_size, void* d_ws, size_t ws_size,
                              hipStream_t stream) {
  const float* x      = (const float*)d_in[0];
  const float* wave_w = (const float*)d_in[1];
  const float* bn_g   = (const float*)d_in[2];
  const float* bn_b   = (const float*)d_in[3];
  const float* miu_ap = (const float*)d_in[4];
  const float* miu_bp = (const float*)d_in[5];
  const float* miu_an = (const float*)d_in[6];
  const float* miu_bn = (const float*)d_in[7];
  const float* ae_w1  = (const float*)d_in[8];
  const float* ae_b1  = (const float*)d_in[9];
  const float* ae_w2  = (const float*)d_in[10];
  const float* ae_b2  = (const float*)d_in[11];
  const float* ae_w3  = (const float*)d_in[12];
  const float* ae_b3  = (const float*)d_in[13];
  const float* ae_w4  = (const float*)d_in[14];
  const float* ae_b4  = (const float*)d_in[15];
  const float* and2_w = (const float*)d_in[16];
  const float* or2_w  = (const float*)d_in[17];
  const float* and_w  = (const float*)d_in[18];
  const float* or_w   = (const float*)d_in[19];
  const float* fc1_w  = (const float*)d_in[20];
  const float* fc1_b  = (const float*)d_in[21];
  const float* fc2_w  = (const float*)d_in[22];
  const float* fc2_b  = (const float*)d_in[23];
  const float* fc3_w  = (const float*)d_in[24];
  const float* fc3_b  = (const float*)d_in[25];

  float* ws = (float*)d_ws;
  float* conv1b = ws + o_conv1;
  float* xf     = ws + o_xf;
  float* xmiu   = ws + o_xmiu;
  float* h1     = ws + o_h1;
  float* h2     = ws + o_h2;
  float* h3     = ws + o_h3;
  float* h4     = ws + o_h4;
  float* inp    = ws + o_inp;
  float* x1     = ws + o_x1;
  float* x2     = ws + o_x2;
  float* wa2    = ws + o_wa2;
  float* wo2    = ws + o_wo2;
  float* wa     = ws + o_wa;
  float* wo     = ws + o_wo;
  float* stat   = ws + o_stat;
  _Float16* apack = (_Float16*)(ws + o_apack);
  _Float16* w1p   = (_Float16*)(ws + o_w1p);
  _Float16* w2p   = (_Float16*)(ws + o_w2p);
  _Float16* w3p   = (_Float16*)(ws + o_w3p);
  _Float16* f1h   = (_Float16*)(ws + o_f1h);
  _Float16* f2h   = (_Float16*)(ws + o_f2h);

  const int TB = 256;
  int n;

  n = Bn * C1 * L1;
  tln_conv1<<<cdiv(n, TB), TB, 0, stream>>>(x, wave_w, conv1b, n);
  tln_bnstats<<<C1, 256, 0, stream>>>(conv1b, stat);
  n = Bn * C1 * L1P;
  tln_bnpool<<<cdiv(n, TB), TB, 0, stream>>>(conv1b, stat, bn_g, bn_b, xf, n);
  n = Bn * C1 * L2;
  tln_miu<<<cdiv(n, TB), TB, 0, stream>>>(xf, miu_ap, miu_bp, miu_an, miu_bn, xmiu, n);
  n = Bn * CA1 * LA1;
  tln_ae1<<<cdiv(n, TB), TB, 0, stream>>>(xmiu, ae_w1, ae_b1, h1, n);
  n = Bn * CA2 * LA2;
  tln_ae2<<<cdiv(n, TB), TB, 0, stream>>>(h1, ae_w2, ae_b2, h2, n);
  n = Bn * C3 * L3;
  tln_ae3<<<cdiv(n, TB), TB, 0, stream>>>(h2, ae_w3, ae_b3, h3, n);
  n = Bn * C4 * L4;
  tln_ae4<<<cdiv(n, TB), TB, 0, stream>>>(h3, ae_w4, ae_b4, h4, n);
  n = Bn * CM * LI;
  tln_seg<<<cdiv(n, TB), TB, 0, stream>>>(h4, xmiu, inp, n);
  n = Bn * CI * LT;
  tln_until<<<cdiv(n, TB), TB, 0, stream>>>(inp, x1, n);
  tln_softmax<<<1, 64, 0, stream>>>(and2_w, or2_w, and_w, or_w, wa2, wo2, wa, wo);
  n = Bn * CL1 * LT;
  tln_logic1<<<cdiv(n, TB), TB, 0, stream>>>(inp, wa2, wo2, x1, n);
  n = Bn * CL2 * LT;
  tln_logic2<<<cdiv(n, TB), TB, 0, stream>>>(x1, wa, wo, x2, n);

  // pack activations + weights to zero-padded f16
  n = Mrows * Kp1;
  tln_pack<<<cdiv(n, TB), TB, 0, stream>>>(x2,    apack, Mrows, K1, Kp1, n);
  n = N1 * Kp1;
  tln_pack<<<cdiv(n, TB), TB, 0, stream>>>(fc1_w, w1p,   N1,    K1, Kp1, n);
  n = N2 * K2;
  tln_pack<<<cdiv(n, TB), TB, 0, stream>>>(fc2_w, w2p,   N2,    K2, K2,  n);
  n = Np3 * K3;
  tln_pack<<<cdiv(n, TB), TB, 0, stream>>>(fc3_w, w3p,   N3,    K3, K3,  n);

  // FC head on the matrix pipes: 8 waves/block, one 16x16 tile per wave
  tln_gemm_f16out<<<(Mrows / 16) * (N1 / 16) / 8, 256, 0, stream>>>(apack, w1p, fc1_b, f1h, Kp1, N1, N1 / 16);
  tln_gemm_f16out<<<(Mrows / 16) * (N2 / 16) / 8, 256, 0, stream>>>(f1h,   w2p, fc2_b, f2h, K2,  N2, N2 / 16);
  tln_gemm_f32out<<<(Mrows / 16) * (Np3 / 16) / 8, 256, 0, stream>>>(f2h,  w3p, fc3_b, (float*)d_out, K3, N3, Np3 / 16);
}
